// Attention_29807073034381
// MI455X (gfx1250) — matrix operand order repaired and checked by reading, not compiled, not run on hardware
//
#include <hip/hip_runtime.h>
#include <stdint.h>

typedef __attribute__((ext_vector_type(16))) __bf16 v16bf;
typedef __attribute__((ext_vector_type(8)))  __bf16 v8bf;
typedef __attribute__((ext_vector_type(8)))  float  v8f;

#define EMBED  1024
#define HEADS  16
#define HDIM   64
#define SEQ    2048
#define BSZ    2
#define TRIPLE (3 * EMBED)
#define LORA   8
#define MROWS  (BSZ * SEQ)   // 4096 token rows

// Wait until this wave's outstanding async(LDS) loads <= n.
#define WAIT_ASYNC(n) asm volatile("s_wait_asynccnt %0" :: "n"(n) : "memory")

// ---------- helpers ----------

static __device__ inline unsigned short f32_to_bf16(float f) {
  unsigned int u = __float_as_uint(f);
  u += 0x7FFFu + ((u >> 16) & 1u);      // round-to-nearest-even
  return (unsigned short)(u >> 16);
}

// Async copy of one 16B chunk: global -> LDS (ASYNCcnt-tracked, GV mode).
static __device__ inline void async_b128(unsigned short* lds_dst,
                                         const unsigned short* gsrc) {
  uint32_t lds_addr = (uint32_t)(uintptr_t)lds_dst;  // generic ptr low bits = LDS offset
  asm volatile("global_load_async_to_lds_b128 %0, %1, off"
               :: "v"(lds_addr), "v"(gsrc) : "memory");
}

// A-matrix fragment (16-bit 16x32, ISA 7.12.2): lanes 0-15 -> K = kk+h (h<8),
// kk+h+8 (h>=8) with kk = 8*(lane>=16). Two contiguous 16B chunks.
static __device__ inline v16bf load_a_frag(const unsigned short* p, int kk) {
  v8bf lo = *(const v8bf*)(p + kk);
  v8bf hi = *(const v8bf*)(p + kk + 16);
  return __builtin_shufflevector(lo, hi, 0,1,2,3,4,5,6,7,8,9,10,11,12,13,14,15);
}

// B-matrix fragment (16-bit 32x16): same K order as the A fragment: K = kk+s
// (s<8), kk+s+8 (s>=8) with kk = boff/2 = 8*(lane>=16). Two 16B chunks.
static __device__ inline v16bf load_b_frag(const unsigned short* p, int boff) {
  v8bf lo = *(const v8bf*)(p + boff / 2);
  v8bf hi = *(const v8bf*)(p + boff / 2 + 16);
  return __builtin_shufflevector(lo, hi, 0,1,2,3,4,5,6,7,8,9,10,11,12,13,14,15);
}

static __device__ inline v8f wmma_bf16(v16bf a, v16bf b, v8f c) {
  return __builtin_amdgcn_wmma_f32_16x16x32_bf16(false, a, false, b, (short)0, c,
                                                 false, false);
}

// ---------- kernel 1: fp32 -> bf16 conversion ----------

__global__ void f32_to_bf16_kernel(const float* __restrict__ src,
                                   unsigned short* __restrict__ dst, int n) {
  int i = blockIdx.x * blockDim.x + threadIdx.x;
  int stride = gridDim.x * blockDim.x;
  for (; i < n; i += stride) dst[i] = f32_to_bf16(src[i]);
}

// ---------- kernel 2: LoRA rank-8 intermediate t = x @ A^T  [4096 x 8] fp32 ----------

__global__ void lora_t_kernel(const float* __restrict__ x,
                              const float* __restrict__ la,
                              float* __restrict__ t) {
  int idx = blockIdx.x * blockDim.x + threadIdx.x;
  if (idx >= MROWS * LORA) return;
  int m = idx >> 3, r = idx & 7;
  const float* xr = x  + (size_t)m * EMBED;
  const float* ar = la + (size_t)r * EMBED;
  float s = 0.f;
  for (int k = 0; k < EMBED; ++k) s += xr[k] * ar[k];
  t[idx] = s;
}

// ---------- shared GEMM core: 128m x 64n block tile, 4 waves, K = EMBED ----------
// Wave w computes rows [m0+w*32, m0+w*32+32) x cols [n0, n0+64).
// B panel (64 rows of W x 32 k) is async-copied to LDS, double-buffered and
// shared by all 4 waves; the ping-pong is written as two identical phases so
// accumulators keep one register home across the unrolled body.

static __device__ inline void gemm_core_128x64(
    const unsigned short* __restrict__ Amat,   // [*, EMBED] bf16, row-major
    const unsigned short* __restrict__ Bmat,   // W rows: [n, EMBED] bf16
    int m0_wave, int n0, unsigned short* lds_b /* [2][64*32] */, v8f acc[2][4]) {
  const int tid  = threadIdx.x;
  const int lane = tid & 31;
  const int l16  = lane & 15, half = lane >> 4;
  const int kk   = half * 8, boff = half * 16;
  const unsigned short* a0 = Amat + (size_t)(m0_wave + l16) * EMBED;
  const unsigned short* a1 = a0 + (size_t)16 * EMBED;
  const unsigned short* Bbase = Bmat + (size_t)n0 * EMBED;
  unsigned short* buf0 = lds_b;
  unsigned short* buf1 = lds_b + 64 * 32;

  // stage one 64x32 B panel (4KB = 256 x 16B chunks, 2 per thread)
  auto stage = [&](int k0, unsigned short* dst) {
#pragma unroll
    for (int it = 0; it < 2; ++it) {
      const int chunk = tid + it * 128;
      const int row = chunk >> 2, q = chunk & 3;
      async_b128(dst + row * 32 + q * 8,
                 Bbase + (size_t)row * EMBED + k0 + q * 8);
    }
  };
  auto compute = [&](int k0, const unsigned short* bbuf) {
    v16bf A0 = load_a_frag(a0 + k0, kk);
    v16bf A1 = load_a_frag(a1 + k0, kk);
    __builtin_prefetch(a0 + k0 + 64, 0, 1);
#pragma unroll
    for (int c = 0; c < 4; ++c) {
      v16bf bfr = load_b_frag(bbuf + (c * 16 + l16) * 32, boff);
      acc[0][c] = wmma_bf16(A0, bfr, acc[0][c]);
      acc[1][c] = wmma_bf16(A1, bfr, acc[1][c]);
    }
  };

  const int NK = EMBED / 32;   // 32 (even)
  stage(0, buf0);
  for (int kt = 0; kt < NK; kt += 2) {
    stage((kt + 1) * 32, buf1);
    WAIT_ASYNC(2);
    __syncthreads();
    compute(kt * 32, buf0);
    __syncthreads();
    if (kt + 2 < NK) { stage((kt + 2) * 32, buf0); WAIT_ASYNC(2); }
    else             { WAIT_ASYNC(0); }
    __syncthreads();
    compute((kt + 1) * 32, buf1);
    __syncthreads();
  }
}

// ---------- kernel 3: QKV GEMM + bias + LoRA epilogue, scatter Q/K/V^T bf16 ----------

__global__ __launch_bounds__(128)
void qkv_gemm_kernel(const unsigned short* __restrict__ xbf,
                     const unsigned short* __restrict__ wbf,
                     const float* __restrict__ bias,
                     const float* __restrict__ tmat,
                     const float* __restrict__ lora_b,
                     unsigned short* __restrict__ qbf,
                     unsigned short* __restrict__ kbf,
                     unsigned short* __restrict__ vtbf) {
  __shared__ unsigned short lds_b[2][64 * 32];
  const int lane = threadIdx.x & 31, w = threadIdx.x >> 5;
  const int l16 = lane & 15, half = lane >> 4;
  const int n0 = blockIdx.x * 64;
  const int m0 = blockIdx.y * 128 + w * 32;

  v8f acc[2][4] = {};
  gemm_core_128x64(xbf, wbf, m0, n0, &lds_b[0][0], acc);

#pragma unroll
  for (int seg = 0; seg < 2; ++seg)
#pragma unroll
    for (int c = 0; c < 4; ++c)
#pragma unroll
      for (int r = 0; r < 8; ++r) {
        const int ml = seg * 16 + r + 8 * half;
        const int m  = m0 + ml;
        const int n  = n0 + c * 16 + l16;
        float v = acc[seg][c][r] + bias[n];
        const float* tr = tmat   + (size_t)m * LORA;
        const float* lb = lora_b + (size_t)n * LORA;
        float d = 0.f;
#pragma unroll
        for (int j = 0; j < LORA; ++j) d += tr[j] * lb[j];
        v += d;
        const int sec = n >> 10;            // 0=Q 1=K 2=V
        const int e   = n & (EMBED - 1);
        const int h   = e >> 6, dd = e & 63;
        const int bb  = m >> 11, s = m & (SEQ - 1);
        const size_t bh = (size_t)bb * HEADS + h;
        if (sec == 0)
          qbf[(bh * SEQ + s) * HDIM + dd] = f32_to_bf16(v * 0.125f);  // 1/sqrt(64)
        else if (sec == 1)
          kbf[(bh * SEQ + s) * HDIM + dd] = f32_to_bf16(v);
        else
          vtbf[(bh * HDIM + dd) * SEQ + s] = f32_to_bf16(v);
      }
}

// ---------- kernel 4: flash attention ----------
// Block = 4 waves x 16 queries, sharing async-staged K/V tiles (32 keys per
// step, double-buffered). Online softmax is fully lane-parallel: each score
// row's 16 columns live in 16 lanes of one half-wave with the SAME lane<->row
// mapping as the O accumulators, so row max/sum are 4 __shfl_xor reductions
// and alpha/1/l rescales stay in registers. LDS is only used for K/V tiles
// and the tiny P transpose (C-layout -> A-fragment layout), which is
// wave-private (no extra barriers).

__global__ __launch_bounds__(128)
void flash_attn_kernel(const unsigned short* __restrict__ qbf,
                       const unsigned short* __restrict__ kbf,
                       const unsigned short* __restrict__ vtbf,
                       const float* __restrict__ mask,
                       unsigned short* __restrict__ obf) {
  __shared__ unsigned short lds_k[2][32 * 64];   // [key][d]
  __shared__ unsigned short lds_v[2][64 * 32];   // [d][key]
  __shared__ unsigned short lds_p[4][16][32];    // per-wave P transpose

  const int tid  = threadIdx.x;
  const int lane = tid & 31, w = tid >> 5;
  const int l16  = lane & 15, half = lane >> 4;
  const int kk   = half * 8, boff = half * 16;
  const int q0   = blockIdx.x * 64;
  const int h    = blockIdx.y, b = blockIdx.z;
  const size_t bh = (size_t)b * HEADS + h;
  const unsigned short* Q  = qbf  + (bh * SEQ + q0 + (size_t)w * 16) * HDIM;
  const unsigned short* Kp = kbf  + bh * SEQ * HDIM;
  const unsigned short* Vt = vtbf + bh * HDIM * SEQ;
  const float* Mrow = mask + ((size_t)b * SEQ + q0 + (size_t)w * 16) * SEQ;

  // persistent Q fragments: 16 queries x K=64 (two K=32 steps)
  v16bf aQ[2];
#pragma unroll
  for (int t = 0; t < 2; ++t) aQ[t] = load_a_frag(Q + l16 * HDIM + t * 32, kk);

  v8f oacc[4] = {};
  float row_m[8], row_l[8];              // stats for rows r + 8*half (per lane)
#pragma unroll
  for (int r = 0; r < 8; ++r) { row_m[r] = -3.0e38f; row_l[r] = 0.f; }

  // stage K tile (contiguous 4KB) + V tile (64 rows x 64B) : 4 chunks/thread
  auto stage = [&](int kb, int buf) {
#pragma unroll
    for (int it = 0; it < 2; ++it) {
      const int chunk = tid + it * 128;
      async_b128(&lds_k[buf][chunk * 8], Kp + (size_t)kb * HDIM + chunk * 8);
      const int row = chunk >> 2, q = chunk & 3;
      async_b128(&lds_v[buf][row * 32 + q * 8],
                 Vt + (size_t)row * SEQ + kb + q * 8);
    }
  };

  auto process = [&](int kb, int buf) {
    // ---- scores: 16q x 32k, K-dim = D = 64, K tile from LDS ----
    v8f sacc[2] = {};
#pragma unroll
    for (int j = 0; j < 2; ++j) {
      const unsigned short* krow = &lds_k[buf][(j * 16 + l16) * 64];
#pragma unroll
      for (int t = 0; t < 2; ++t)
        sacc[j] = wmma_bf16(aQ[t], load_b_frag(krow + t * 32, boff), sacc[j]);
    }
    // ---- lane-parallel online softmax, one row (r + 8*half) per register ----
#pragma unroll
    for (int r = 0; r < 8; ++r) {
      const int ml = r + 8 * half;
      const float s0 = sacc[0][r] + Mrow[(size_t)ml * SEQ + kb + l16];
      const float s1 = sacc[1][r] + Mrow[(size_t)ml * SEQ + kb + 16 + l16];
      float t = fmaxf(s0, s1);
#pragma unroll
      for (int msk = 8; msk >= 1; msk >>= 1)
        t = fmaxf(t, __shfl_xor(t, msk, 16));
      const float mnew  = fmaxf(row_m[r], t);
      const float alpha = __expf(row_m[r] - mnew);
      row_m[r] = mnew;
      const float p0 = __expf(s0 - mnew);
      const float p1 = __expf(s1 - mnew);
      float u = p0 + p1;
#pragma unroll
      for (int msk = 8; msk >= 1; msk >>= 1)
        u += __shfl_xor(u, msk, 16);
      row_l[r] = row_l[r] * alpha + u;
#pragma unroll
      for (int c = 0; c < 4; ++c) oacc[c][r] *= alpha;
      lds_p[w][ml][l16]      = f32_to_bf16(p0);
      lds_p[w][ml][16 + l16] = f32_to_bf16(p1);
    }
    // ---- P @ V (K-dim = 32 keys), P re-read in A-fragment layout ----
    v16bf aP = load_a_frag(&lds_p[w][l16][0], kk);
#pragma unroll
    for (int c = 0; c < 4; ++c)
      oacc[c] = wmma_bf16(aP, load_b_frag(&lds_v[buf][(c * 16 + l16) * 32], boff),
                          oacc[c]);
  };

  const int NT = SEQ / 32;   // 64 (even)
  stage(0, 0);
  for (int kt = 0; kt < NT; kt += 2) {
    stage((kt + 1) * 32, 1);
    WAIT_ASYNC(4);
    __syncthreads();
    process(kt * 32, 0);
    __syncthreads();
    if (kt + 2 < NT) { stage((kt + 2) * 32, 0); WAIT_ASYNC(4); }
    else             { WAIT_ASYNC(0); }
    __syncthreads();
    process((kt + 1) * 32, 1);
    __syncthreads();
  }

  // ---- normalize from registers, write bf16 [B, S, E] for the out-proj ----
#pragma unroll
  for (int c = 0; c < 4; ++c)
#pragma unroll
    for (int r = 0; r < 8; ++r) {
      const int ml = r + 8 * half;
      const float v = oacc[c][r] / row_l[r];
      const int s = q0 + w * 16 + ml;
      obf[((size_t)b * SEQ + s) * EMBED + h * HDIM + c * 16 + l16] = f32_to_bf16(v);
    }
}

// ---------- kernel 5: out-projection GEMM + bias, fp32 output ----------

__global__ __launch_bounds__(128)
void outproj_gemm_kernel(const unsigned short* __restrict__ obf,
                         const unsigned short* __restrict__ wbf,
                         const float* __restrict__ bias,
                         float* __restrict__ out) {
  __shared__ unsigned short lds_b[2][64 * 32];
  const int lane = threadIdx.x & 31, w = threadIdx.x >> 5;
  const int l16 = lane & 15, half = lane >> 4;
  const int n0 = blockIdx.x * 64;
  const int m0 = blockIdx.y * 128 + w * 32;

  v8f acc[2][4] = {};
  gemm_core_128x64(obf, wbf, m0, n0, &lds_b[0][0], acc);

#pragma unroll
  for (int seg = 0; seg < 2; ++seg)
#pragma unroll
    for (int c = 0; c < 4; ++c)
#pragma unroll
      for (int r = 0; r < 8; ++r) {
        const int ml = seg * 16 + r + 8 * half;
        const int n  = n0 + c * 16 + l16;
        out[(size_t)(m0 + ml) * EMBED + n] = acc[seg][c][r] + bias[n];
      }
}

// ---------- host ----------

extern "C" void kernel_launch(void* const* d_in, const int* in_sizes, int n_in,
                              void* d_out, int out_size, void* d_ws, size_t ws_size,
                              hipStream_t stream) {
  (void)in_sizes; (void)n_in; (void)out_size; (void)ws_size;
  const float* x      = (const float*)d_in[0];
  const float* mask   = (const float*)d_in[1];
  const float* w_in   = (const float*)d_in[2];
  const float* b_in   = (const float*)d_in[3];
  const float* w_out  = (const float*)d_in[4];
  const float* b_out  = (const float*)d_in[5];
  const float* lora_a = (const float*)d_in[6];
  const float* lora_b = (const float*)d_in[7];
  float* out = (float*)d_out;

  char* ws = (char*)d_ws;
  size_t off = 0;
  auto alloc = [&](size_t bytes) -> char* {
    char* p = ws + off;
    off += (bytes + 255) & ~(size_t)255;
    return p;
  };
  unsigned short* xbf   = (unsigned short*)alloc((size_t)MROWS * EMBED * 2);   // 8 MB
  unsigned short* winb  = (unsigned short*)alloc((size_t)TRIPLE * EMBED * 2);  // 6 MB
  unsigned short* woutb = (unsigned short*)alloc((size_t)EMBED * EMBED * 2);   // 2 MB
  float*          tmat  = (float*)alloc((size_t)MROWS * LORA * 4);             // 128 KB
  unsigned short* qbf   = (unsigned short*)alloc((size_t)MROWS * EMBED * 2);   // 8 MB
  unsigned short* kbf   = (unsigned short*)alloc((size_t)MROWS * EMBED * 2);   // 8 MB
  unsigned short* vtbf  = (unsigned short*)alloc((size_t)MROWS * EMBED * 2);   // 8 MB
  unsigned short* obf   = (unsigned short*)alloc((size_t)MROWS * EMBED * 2);   // 8 MB

  f32_to_bf16_kernel<<<1024, 256, 0, stream>>>(x, xbf, MROWS * EMBED);
  f32_to_bf16_kernel<<<1024, 256, 0, stream>>>(w_in, winb, TRIPLE * EMBED);
  f32_to_bf16_kernel<<<512, 256, 0, stream>>>(w_out, woutb, EMBED * EMBED);
  lora_t_kernel<<<(MROWS * LORA + 255) / 256, 256, 0, stream>>>(x, lora_a, tmat);

  qkv_gemm_kernel<<<dim3(TRIPLE / 64, MROWS / 128), 128, 0, stream>>>(
      xbf, winb, b_in, tmat, lora_b, qbf, kbf, vtbf);

  flash_attn_kernel<<<dim3(SEQ / 64, HEADS, BSZ), 128, 0, stream>>>(
      qbf, kbf, vtbf, mask, obf);

  outproj_gemm_kernel<<<dim3(EMBED / 64, MROWS / 128), 128, 0, stream>>>(
      obf, woutb, b_out, out);
}
